// Seq2Seq_58239756534225
// MI455X (gfx1250) — matrix operand order repaired and checked by reading, not compiled, not run on hardware
//
#include <hip/hip_runtime.h>
#include <hip/hip_bf16.h>
#include <cstddef>

typedef __bf16 bf16;
typedef __attribute__((ext_vector_type(16))) __bf16 v16bf;
typedef __attribute__((ext_vector_type(8)))  __bf16 v8bf;
typedef __attribute__((ext_vector_type(8)))  float   v8f;

#define Bsz     256
#define SRCLEN  256
#define Ddim    64
#define Hdim    512
#define TGT     64
#define FOURH   2048
#define NTILES  128   // 4H / 16

// ---------------- utility kernels ----------------
__global__ void cvt_f32_bf16(const float* __restrict__ in, bf16* __restrict__ out, int n) {
    int i = blockIdx.x * blockDim.x + threadIdx.x;
    if (i < n) out[i] = (bf16)in[i];
}
__global__ void zero_bf16(bf16* __restrict__ p, int n) {
    int i = blockIdx.x * blockDim.x + threadIdx.x;
    if (i < n) p[i] = (bf16)0.0f;
}
__global__ void zero_f32(float* __restrict__ p, int n) {
    int i = blockIdx.x * blockDim.x + threadIdx.x;
    if (i < n) p[i] = 0.0f;
}
__global__ void add_vec(const float* __restrict__ a, const float* __restrict__ b,
                        float* __restrict__ o, int n) {
    int i = blockIdx.x * blockDim.x + threadIdx.x;
    if (i < n) o[i] = a[i] + b[i];
}

// Pack W = [Wih ; Whh] (concatenated along K) into WMMA B-fragment order:
// dst[((kt*nTiles + ntile)*32 + lane)*16 + i] holds K = kt*32 + (lane>=16?8:0) + i + (i>=8?8:0),
// N = ntile*16 + (lane&15).  (lanes 0-15 K=0..7,16..23 N=lane;
// lanes 16-31 K=8..15,24..31 N=lane-16; matches A fragment order.)
__global__ void pack_weights(const float* __restrict__ Wih, const float* __restrict__ Whh,
                             int idim, int Kdim, int nTiles, bf16* __restrict__ dst) {
    int tid = blockIdx.x * blockDim.x + threadIdx.x;
    int total = (Kdim >> 5) * nTiles * 32;
    if (tid >= total) return;
    int lane  = tid & 31;
    int ntile = (tid >> 5) % nTiles;
    int kt    = tid / (32 * nTiles);
    int n     = ntile * 16 + (lane & 15);
    int khalf = (lane >> 4) * 8;
    bf16* out = dst + (size_t)tid * 16;
    #pragma unroll
    for (int i = 0; i < 16; ++i) {
        int k = (kt << 5) + khalf + i + (i >= 8 ? 8 : 0);
        float v = (k < idim) ? Wih[(size_t)n * idim + k]
                             : Whh[(size_t)n * Hdim + (k - idim)];
        out[i] = (bf16)v;
    }
}

// ---------------- fused LSTM cell ----------------
// gates = [x|h] @ [Wih;Whh]^T + bias ; i,f,o=sigmoid, g=tanh ;
// c = f*c + i*tanh(g) ; h_out = o*tanh(c)  (bf16, ping-pong buffer)
// grid (8, 16): blockIdx.x = M-tile of 32 rows, blockIdx.y = 32-col slice of each gate.
// Wave (gate, nt): loads ONE B fragment per k-tile, reuses it across both 16-row
// M-tiles (halves L2 weight traffic).  LDS A is double-buffered: next 64-wide K
// chunk is fetched into registers while WMMAs consume the current buffer.
__global__ __launch_bounds__(256) void lstm_cell(
    const bf16* __restrict__ X, int xStride, int idim,
    const bf16* __restrict__ Hin, bf16* __restrict__ Hout,
    float* __restrict__ Cst,
    const bf16* __restrict__ Wp, int Kdim,
    const float* __restrict__ bias)
{
    __shared__ bf16  ldsA[2][32][72];     // double-buffered 32 rows x 64 K (+pad)
    __shared__ float ldsG[4][32][32];     // 4 gate tiles for epilogue exchange

    const int tid  = threadIdx.x;
    const int lane = tid & 31;
    const int wid  = tid >> 5;            // 8 waves
    const int bm   = blockIdx.x;
    const int bn   = blockIdx.y;
    const int gate = wid >> 1;            // 0..3
    const int nt   = wid & 1;             // 16-col half of the 32-col slice

    v8f accm0 = {}; v8f accm1 = {};

    const int row   = tid >> 3;           // 0..31 (coop A load)
    const int colq  = (tid & 7) * 8;      // 0..56
    const int bglob = bm * 32 + row;

    const int nKk = Kdim >> 6;            // 64-wide K chunks (idim multiple of 64)

    // stage chunk 0
    {
        const int kc = colq;
        const bf16* s = (kc < idim) ? X + (size_t)bglob * xStride + kc
                                    : Hin + (size_t)bglob * Hdim + (kc - idim);
        *(v8bf*)&ldsA[0][row][colq] = *(const v8bf*)s;
    }
    __syncthreads();

    const int ntg = gate * 32 + bn * 2 + nt;
    const int kh  = (lane >> 4) * 8;
    const int arl = lane & 15;

    for (int kk = 0; kk < nKk; ++kk) {
        const int cb = kk & 1;
        // prefetch next chunk into registers (overlaps with WMMA below)
        v8bf nxt = {};
        const bool more = (kk + 1) < nKk;
        if (more) {
            const int kc = ((kk + 1) << 6) + colq;
            const bf16* s = (kc < idim) ? X + (size_t)bglob * xStride + kc
                                        : Hin + (size_t)bglob * Hdim + (kc - idim);
            nxt = *(const v8bf*)s;
        }
        #pragma unroll
        for (int ks = 0; ks < 2; ++ks) {
            const int kt = kk * 2 + ks;
            // A fragments (16x32 bf16, ISA layout): lanes<16 -> K 0..7 & 16..23,
            // lanes>=16 -> K 8..15 & 24..31, row M = lane&15.
            v8bf lo0 = *(const v8bf*)&ldsA[cb][arl][ks * 32 + kh];
            v8bf hi0 = *(const v8bf*)&ldsA[cb][arl][ks * 32 + 16 + kh];
            v8bf lo1 = *(const v8bf*)&ldsA[cb][16 + arl][ks * 32 + kh];
            v8bf hi1 = *(const v8bf*)&ldsA[cb][16 + arl][ks * 32 + 16 + kh];
            v16bf a0 = __builtin_shufflevector(lo0, hi0,
                        0,1,2,3,4,5,6,7,8,9,10,11,12,13,14,15);
            v16bf a1 = __builtin_shufflevector(lo1, hi1,
                        0,1,2,3,4,5,6,7,8,9,10,11,12,13,14,15);
            v16bf b = *(const v16bf*)(Wp + ((size_t)(kt * NTILES + ntg) * 32 + lane) * 16);
            accm0 = __builtin_amdgcn_wmma_f32_16x16x32_bf16(false, a0, false, b,
                                                            (short)0, accm0, false, false);
            accm1 = __builtin_amdgcn_wmma_f32_16x16x32_bf16(false, a1, false, b,
                                                            (short)0, accm1, false, false);
        }
        if (more) *(v8bf*)&ldsA[1 - cb][row][colq] = nxt;
        __syncthreads();
    }

    // dump accumulators to LDS (C layout: VGPR v -> M = v + 8*(lane>=16), N = lane&15)
    {
        const int mhalf = (lane >> 4) << 3;
        const int n0 = nt * 16 + (lane & 15);
        #pragma unroll
        for (int v = 0; v < 8; ++v) {
            ldsG[gate][mhalf + v][n0]      = accm0[v];
            ldsG[gate][16 + mhalf + v][n0] = accm1[v];
        }
    }
    __syncthreads();

    // fused LSTM epilogue
    for (int e = tid; e < 32 * 32; e += 256) {
        const int m = e >> 5, n = e & 31;
        const int b = bm * 32 + m;
        const int j = bn * 32 + n;
        float xi = ldsG[0][m][n] + bias[j];
        float xf = ldsG[1][m][n] + bias[512  + j];
        float xg = ldsG[2][m][n] + bias[1024 + j];
        float xo = ldsG[3][m][n] + bias[1536 + j];
        float si = 1.0f / (1.0f + __expf(-xi));
        float sf = 1.0f / (1.0f + __expf(-xf));
        float so = 1.0f / (1.0f + __expf(-xo));
        const size_t idx = (size_t)b * Hdim + j;
        float cn = sf * Cst[idx] + si * tanhf(xg);
        Cst[idx]  = cn;
        Hout[idx] = (bf16)(so * tanhf(cn));
    }
}

// ---------------- decoder FC head: pred = h @ fcW^T + b ----------------
// grid (8,2), 128 threads; writes f32 into d_out[:,t,:] and bf16 next-step input.
__global__ __launch_bounds__(128) void fc_gemm(
    const bf16* __restrict__ X, const bf16* __restrict__ Wp,
    const float* __restrict__ bias,
    float* __restrict__ outF, bf16* __restrict__ predB)
{
    __shared__ bf16 ldsA[32][72];
    const int tid  = threadIdx.x;
    const int lane = tid & 31;
    const int wid  = tid >> 5;            // 4 waves
    const int bm   = blockIdx.x, bn = blockIdx.y;
    const int mt   = wid & 1, nt = wid >> 1;
    v8f acc = {};

    const int row   = tid >> 2;           // 0..31
    const int colq  = (tid & 3) * 16;     // 0,16,32,48
    const int bglob = bm * 32 + row;

    for (int kk = 0; kk < (Hdim >> 6); ++kk) {
        const bf16* s = X + (size_t)bglob * Hdim + (kk << 6) + colq;
        *(v8bf*)&ldsA[row][colq]     = *(const v8bf*)s;
        *(v8bf*)&ldsA[row][colq + 8] = *(const v8bf*)(s + 8);
        __syncthreads();
        #pragma unroll
        for (int ks = 0; ks < 2; ++ks) {
            const int kt = kk * 2 + ks;
            const int ar = mt * 16 + (lane & 15);
            const int kh = (lane >> 4) * 8;
            v8bf lo = *(const v8bf*)&ldsA[ar][ks * 32 + kh];
            v8bf hi = *(const v8bf*)&ldsA[ar][ks * 32 + 16 + kh];
            v16bf a = __builtin_shufflevector(lo, hi,
                       0,1,2,3,4,5,6,7,8,9,10,11,12,13,14,15);
            const int ntg = bn * 2 + nt;  // 4 N-tiles total
            v16bf b = *(const v16bf*)(Wp + ((size_t)(kt * 4 + ntg) * 32 + lane) * 16);
            acc = __builtin_amdgcn_wmma_f32_16x16x32_bf16(false, a, false, b,
                                                          (short)0, acc, false, false);
        }
        __syncthreads();
    }
    const int mbase = bm * 32 + mt * 16 + ((lane >> 4) << 3);
    const int n = bn * 32 + nt * 16 + (lane & 15);
    #pragma unroll
    for (int v = 0; v < 8; ++v) {
        const int b = mbase + v;
        float val = acc[v] + bias[n];
        outF[(size_t)b * (TGT * Ddim) + n] = val;   // outF pre-offset by t*Ddim
        predB[(size_t)b * Ddim + n] = (bf16)val;
    }
}

// ---------------- host orchestration ----------------
static inline int cdiv(int a, int b) { return (a + b - 1) / b; }

extern "C" void kernel_launch(void* const* d_in, const int* in_sizes, int n_in,
                              void* d_out, int out_size, void* d_ws, size_t ws_size,
                              hipStream_t stream) {
    (void)in_sizes; (void)n_in; (void)out_size; (void)ws_size;
    const float* src = (const float*)d_in[0];
    const float* W[4][4]; // [enc0,enc1,dec0,dec1][Wih,Whh,bih,bhh]
    for (int l = 0; l < 4; ++l)
        for (int p = 0; p < 4; ++p)
            W[l][p] = (const float*)d_in[1 + l * 4 + p];
    const float* fcW = (const float*)d_in[17];
    const float* fcB = (const float*)d_in[18];
    float* out = (float*)d_out;

    // workspace carve-out
    char* ws = (char*)d_ws;
    size_t off = 0;
    auto alloc = [&](size_t bytes) -> char* {
        char* p = ws + off;
        off += (bytes + 255) & ~(size_t)255;
        return p;
    };
    const size_t kt576  = 576 / 32, kt1024 = 1024 / 32, kt512 = 512 / 32;
    bf16* pwE0 = (bf16*)alloc(kt576  * NTILES * 32 * 16 * sizeof(bf16));
    bf16* pwE1 = (bf16*)alloc(kt1024 * NTILES * 32 * 16 * sizeof(bf16));
    bf16* pwD0 = (bf16*)alloc(kt576  * NTILES * 32 * 16 * sizeof(bf16));
    bf16* pwD1 = (bf16*)alloc(kt1024 * NTILES * 32 * 16 * sizeof(bf16));
    bf16* pwFC = (bf16*)alloc(kt512  * 4      * 32 * 16 * sizeof(bf16));
    float* bs[4];
    for (int l = 0; l < 4; ++l) bs[l] = (float*)alloc(FOURH * sizeof(float));
    bf16* srcbf = (bf16*)alloc((size_t)Bsz * SRCLEN * Ddim * sizeof(bf16));
    bf16* hbuf  = (bf16*)alloc((size_t)2 * 2 * Bsz * Hdim * sizeof(bf16));
    float* cbuf = (float*)alloc((size_t)2 * Bsz * Hdim * sizeof(float));
    bf16* predbf = (bf16*)alloc((size_t)Bsz * Ddim * sizeof(bf16));

    const size_t HS = (size_t)Bsz * Hdim;
    bf16* h[2][2] = { { hbuf, hbuf + HS }, { hbuf + 2 * HS, hbuf + 3 * HS } };
    float* c[2]   = { cbuf, cbuf + HS };
    bf16* pw[4]   = { pwE0, pwE1, pwD0, pwD1 };
    const int Ks[4] = { 576, 1024, 576, 1024 };
    const int id[4] = { 64, 512, 64, 512 };

    // ---- prep ----
    {
        int n = Bsz * SRCLEN * Ddim;
        cvt_f32_bf16<<<cdiv(n, 256), 256, 0, stream>>>(src, srcbf, n);
    }
    for (int l = 0; l < 4; ++l) {
        int total = (Ks[l] >> 5) * NTILES * 32;
        pack_weights<<<cdiv(total, 256), 256, 0, stream>>>(W[l][0], W[l][1], id[l], Ks[l],
                                                           NTILES, pw[l]);
        add_vec<<<cdiv(FOURH, 256), 256, 0, stream>>>(W[l][2], W[l][3], bs[l], FOURH);
    }
    {
        int total = (512 >> 5) * 4 * 32;
        pack_weights<<<cdiv(total, 256), 256, 0, stream>>>(fcW, fcW, 512, 512, 4, pwFC);
    }
    zero_bf16<<<cdiv((int)(4 * HS), 256), 256, 0, stream>>>(hbuf, (int)(4 * HS));
    zero_f32 <<<cdiv((int)(2 * HS), 256), 256, 0, stream>>>(cbuf, (int)(2 * HS));
    zero_bf16<<<cdiv(Bsz * Ddim, 256), 256, 0, stream>>>(predbf, Bsz * Ddim);

    dim3 cellGrid(8, 16);
    int cur0 = 0, cur1 = 0;

    // ---- encoder: 256 steps, 2 stacked layers per step ----
    for (int t = 0; t < SRCLEN; ++t) {
        lstm_cell<<<cellGrid, 256, 0, stream>>>(
            srcbf + (size_t)t * Ddim, SRCLEN * Ddim, 64,
            h[0][cur0], h[0][1 - cur0], c[0], pwE0, 576, bs[0]);
        cur0 ^= 1;
        lstm_cell<<<cellGrid, 256, 0, stream>>>(
            h[0][cur0], Hdim, 512,
            h[1][cur1], h[1][1 - cur1], c[1], pwE1, 1024, bs[1]);
        cur1 ^= 1;
    }

    // ---- decoder: 64 autoregressive steps (states continue from encoder) ----
    for (int t = 0; t < TGT; ++t) {
        lstm_cell<<<cellGrid, 256, 0, stream>>>(
            predbf, Ddim, 64,
            h[0][cur0], h[0][1 - cur0], c[0], pwD0, 576, bs[2]);
        cur0 ^= 1;
        lstm_cell<<<cellGrid, 256, 0, stream>>>(
            h[0][cur0], Hdim, 512,
            h[1][cur1], h[1][1 - cur1], c[1], pwD1, 1024, bs[3]);
        cur1 ^= 1;
        fc_gemm<<<dim3(8, 2), 128, 0, stream>>>(
            h[1][cur1], pwFC, fcB, out + (size_t)t * Ddim, predbf);
    }
}
